// EGNNLayer_31714038514064
// MI455X (gfx1250) — matrix operand order repaired and checked by reading, not compiled, not run on hardware
//
#include <hip/hip_runtime.h>

typedef __attribute__((ext_vector_type(16))) __bf16 v16bf;
typedef __attribute__((ext_vector_type(8)))  float  v8f;

#define NF_DIM 160

// ---------------- WMMA helpers ----------------

static __device__ __forceinline__ v8f wmma_bf16(v16bf a, v16bf b, v8f c) {
  // D(16x16,f32) = A(16x32,bf16) * B(32x16,bf16) + C
  return __builtin_amdgcn_wmma_f32_16x16x32_bf16(false, a, false, b, (short)0, c,
                                                 false, false);
}

static __device__ __forceinline__ v8f vzero() {
  v8f c = {0.f, 0.f, 0.f, 0.f, 0.f, 0.f, 0.f, 0.f};
  return c;
}

// B fragments are stored pre-swizzled: fragment (kc,nc) occupies 512 bf16
// (32 lanes x 16 elems, contiguous per lane). Lane L, elem j holds
// W[kc*32 + (L/16)*8 + (j/8)*16 + (j%8)][nc*16 + (L%16)].
static __device__ __forceinline__ v16bf ldB(const __bf16* __restrict__ base,
                                            int NC, int kc, int nc, int lane) {
  return *(const v16bf*)(base + (((kc * NC + nc) << 5) + lane) * 16);
}

// A-fragment K index for 16-bit A (16x32) layout: lane L, elem j ->
// K = (L/16)*8 + (j/8)*16 + (j%8), M = L%16.
static __device__ __forceinline__ int aK(int half, int j) {
  return half * 8 + ((j >> 3) << 4) + (j & 7);
}

// ---------------- workspace layout (bf16 element offsets) ----------------
#define OFF_WMSS 0        // 64x64   KC=2  NC=4
#define OFF_LMS  4096     // 64x64   KC=2  NC=4
#define OFF_WMVV 8192     // 32x64   KC=1  NC=4
#define OFF_WMSV 10240    // 64x32   KC=2  NC=2
#define OFF_WMVS 12288    // 32x32   KC=1  NC=2
#define OFF_LMV  13312    // 32x32   KC=1  NC=2
#define OFF_WUSS 14336    // 4096x64 KC=128 NC=4
#define OFF_WUVV 276480   // 1024x64 KC=32 NC=4
#define OFF_WUSV 342016   // 2048x32 KC=64 NC=2
#define OFF_WUVS 407552   // 2048x32 KC=64 NC=2
#define OFF_LUS  473088   // 64x64   KC=2  NC=4
#define OFF_LUV  477184   // 32x32   KC=1  NC=2

// ---------------- prep: f32 row-major (KxN) -> scaled bf16 B-fragments ----
__global__ void prep_weight_kernel(const float* __restrict__ src,
                                   __bf16* __restrict__ dst,
                                   int KC, int NC, int Ntot, float scale) {
  int total = KC * NC * 512;
  for (int idx = blockIdx.x * blockDim.x + threadIdx.x; idx < total;
       idx += gridDim.x * blockDim.x) {
    int j    = idx & 15;
    int lane = (idx >> 4) & 31;
    int fid  = idx >> 9;
    int nc   = fid % NC;
    int kc   = fid / NC;
    int k = kc * 32 + ((lane >> 4) << 3) + ((j >> 3) << 4) + (j & 7);
    int n = nc * 16 + (lane & 15);
    dst[idx] = (__bf16)(src[k * Ntot + n] * scale);
  }
}

__global__ void zero_kernel(float4* __restrict__ p, int n4) {
  float4 z = make_float4(0.f, 0.f, 0.f, 0.f);
  for (int i = blockIdx.x * blockDim.x + threadIdx.x; i < n4;
       i += gridDim.x * blockDim.x)
    p[i] = z;
}

// ---------------- edge / message kernel ----------------
// One wave handles 16 edges: A = per-edge features, B = shared weights.
#define EW 4
__global__ __launch_bounds__(EW * 32) void edge_kernel(
    const float* __restrict__ nf, const int* __restrict__ ei,
    const float* __restrict__ ea, float* __restrict__ agg,
    const __bf16* __restrict__ wb, int En) {
  __shared__ __align__(16) float lds_x[EW][16][NF_DIM];
  __shared__ __align__(16) float lds_rd[EW][16][4];
  __shared__ int lds_row[EW][16];
  __shared__ int lds_col[EW][16];

  const int w = threadIdx.x >> 5;
  const int L = threadIdx.x & 31;
  const int tile = blockIdx.x * EW + w;
  const int e0 = tile * 16;
  const bool active = (e0 < En);

  if (active && L < 16) {
    int e = e0 + L;
    lds_row[w][L] = ei[e];       // destination (segment id)
    lds_col[w][L] = ei[En + e];  // source node to gather
    float4 a4 = ((const float4*)ea)[e];
    *(float4*)&lds_rd[w][L][0] = a4;
  }
  __syncthreads();
  if (active) {
#pragma unroll
    for (int e = 0; e < 16; ++e) {
      const float4* src = (const float4*)(nf + (long)lds_col[w][e] * NF_DIM);
      float4* dst = (float4*)&lds_x[w][e][0];
      for (int t = L; t < NF_DIM / 4; t += 32) dst[t] = src[t];
    }
  }
  __syncthreads();
  if (!active) return;

  const int M = L & 15;
  const int half = L >> 4;
  const float* __restrict__ xr = &lds_x[w][M][0];
  const float dist = lds_rd[w][M][3];
  const float r0 = lds_rd[w][M][0], r1 = lds_rd[w][M][1], r2 = lds_rd[w][M][2];

  // Build A fragments (per-lane row M, K per 16-bit A-matrix layout).
  v16bf sA[2], sdA[2], dvA, vA[3], vdA[3];
#pragma unroll
  for (int kc = 0; kc < 2; ++kc) {
#pragma unroll
    for (int j = 0; j < 16; ++j) {
      int k = kc * 32 + aK(half, j);
      float s = xr[k];
      sA[kc][j]  = (__bf16)s;
      sdA[kc][j] = (__bf16)(s * dist);  // dist folded into A (row scalar)
    }
  }
#pragma unroll
  for (int j = 0; j < 16; ++j) {
    int u = aK(half, j);  // 0..31
    const float* vp = xr + 64 + u * 3;
    dvA[j] = (__bf16)(vp[0] * r0 + vp[1] * r1 + vp[2] * r2);  // v . r_vec
#pragma unroll
    for (int i = 0; i < 3; ++i) {
      float vv = vp[i];
      vA[i][j]  = (__bf16)vv;
      vdA[i][j] = (__bf16)(vv * dist);
    }
  }

  const __bf16* Wmss = wb + OFF_WMSS;
  const __bf16* Lms  = wb + OFF_LMS;
  const __bf16* Wmvv = wb + OFF_WMVV;
  const __bf16* Wmsv = wb + OFF_WMSV;
  const __bf16* Wmvs = wb + OFF_WMVS;
  const __bf16* Lmv  = wb + OFF_LMV;

  // m0 (16x64) = (s*dist)@Wm_ss' + s@Lm_s' + (v.r)@Wm_vv'   (scales folded)
  v8f m0[4];
#pragma unroll
  for (int nc = 0; nc < 4; ++nc) {
    v8f c = vzero();
    c = wmma_bf16(sdA[0], ldB(Wmss, 4, 0, nc, L), c);
    c = wmma_bf16(sdA[1], ldB(Wmss, 4, 1, nc, L), c);
    c = wmma_bf16(sA[0],  ldB(Lms,  4, 0, nc, L), c);
    c = wmma_bf16(sA[1],  ldB(Lms,  4, 1, nc, L), c);
    c = wmma_bf16(dvA,    ldB(Wmvv, 4, 0, nc, L), c);
    m0[nc] = c;
  }
  // sv (16x32) = s @ Wm_sv'   (x r_vec applied below)
  v8f svc[2];
#pragma unroll
  for (int nc = 0; nc < 2; ++nc) {
    v8f c = vzero();
    c = wmma_bf16(sA[0], ldB(Wmsv, 2, 0, nc, L), c);
    c = wmma_bf16(sA[1], ldB(Wmsv, 2, 1, nc, L), c);
    svc[nc] = c;
  }
  // m1_i (16x32) = (v_i*dist)@Wm_vs' + v_i@Lm_v' + sv * r_i
  v8f m1[3][2];
#pragma unroll
  for (int i = 0; i < 3; ++i) {
#pragma unroll
    for (int nc = 0; nc < 2; ++nc) {
      v8f c = vzero();
      c = wmma_bf16(vdA[i], ldB(Wmvs, 2, 0, nc, L), c);
      c = wmma_bf16(vA[i],  ldB(Lmv,  2, 0, nc, L), c);
      m1[i][nc] = c;
    }
  }
  // C/D layout: elem j -> row m = half*8 + j, col = L%16
#pragma unroll
  for (int nc = 0; nc < 2; ++nc) {
#pragma unroll
    for (int j = 0; j < 8; ++j) {
      int m = half * 8 + j;
      float sv = svc[nc][j];
      m1[0][nc][j] += sv * lds_rd[w][m][0];
      m1[1][nc][j] += sv * lds_rd[w][m][1];
      m1[2][nc][j] += sv * lds_rd[w][m][2];
    }
  }

  // segment-sum scatter (agg lives in L2: 12.8 MB)
  const int ncol = L & 15;
#pragma unroll
  for (int nc = 0; nc < 4; ++nc) {
    int ch = nc * 16 + ncol;
#pragma unroll
    for (int j = 0; j < 8; ++j) {
      int m = half * 8 + j;
      atomicAdd(&agg[(long)lds_row[w][m] * NF_DIM + ch], m0[nc][j]);
    }
  }
#pragma unroll
  for (int i = 0; i < 3; ++i) {
#pragma unroll
    for (int nc = 0; nc < 2; ++nc) {
      int ch = 64 + (nc * 16 + ncol) * 3 + i;
#pragma unroll
      for (int j = 0; j < 8; ++j) {
        int m = half * 8 + j;
        atomicAdd(&agg[(long)lds_row[w][m] * NF_DIM + ch], m1[i][nc][j]);
      }
    }
  }
}

// ---------------- node / update kernel ----------------
// One wave = 16 nodes. Bilinear contractions done as (outer-product A) @ W
// with K up to 4096, all on the bf16 WMMA pipe.
#define NW 2
__global__ __launch_bounds__(NW * 32) void node_kernel(
    const float* __restrict__ nf, const float* __restrict__ agg,
    const __bf16* __restrict__ wb, float* __restrict__ out, int Nn) {
  __shared__ __align__(16) float lds_x[NW][16][NF_DIM];
  __shared__ __align__(16) float lds_a[NW][16][NF_DIM];

  const int w = threadIdx.x >> 5;
  const int L = threadIdx.x & 31;
  const int tile = blockIdx.x * NW + w;
  const long n0 = (long)tile * 16;
  const bool active = (n0 < Nn);

  if (active) {
#pragma unroll
    for (int m = 0; m < 16; ++m) {
      const float4* s1 = (const float4*)(nf + (n0 + m) * NF_DIM);
      const float4* s2 = (const float4*)(agg + (n0 + m) * NF_DIM);
      float4* d1 = (float4*)&lds_x[w][m][0];
      float4* d2 = (float4*)&lds_a[w][m][0];
      for (int t = L; t < NF_DIM / 4; t += 32) {
        d1[t] = s1[t];
        d2[t] = s2[t];
      }
    }
  }
  __syncthreads();
  if (!active) return;

  const int M = L & 15;
  const int half = L >> 4;
  const float* __restrict__ x = &lds_x[w][M][0];  // xs: [0,64)  xv: 64+u*3+i
  const float* __restrict__ a = &lds_a[w][M][0];  // a0: [0,64)  a1: 64+v*3+i

  const __bf16* Wuss = wb + OFF_WUSS;
  const __bf16* Wuvv = wb + OFF_WUVV;
  const __bf16* Wusv = wb + OFF_WUSV;
  const __bf16* Wuvs = wb + OFF_WUVS;
  const __bf16* Lus  = wb + OFF_LUS;
  const __bf16* Luv  = wb + OFF_LUV;

  // ---------- u0 ----------
  v8f u0c[4];
#pragma unroll
  for (int nc = 0; nc < 4; ++nc) u0c[nc] = vzero();

  // (xs (x) a0)(16x4096) @ Wu_ss'(4096x64):  k = u*64 + v
  for (int kc = 0; kc < 128; ++kc) {
    const int u = kc >> 1;
    const int vb = (kc & 1) << 5;
    const float xu = x[u];
    v16bf afr;
#pragma unroll
    for (int j = 0; j < 16; ++j) {
      int v = vb + aK(half, j);
      afr[j] = (__bf16)(xu * a[v]);
    }
#pragma unroll
    for (int nc = 0; nc < 4; ++nc)
      u0c[nc] = wmma_bf16(afr, ldB(Wuss, 4, kc, nc, L), u0c[nc]);
  }
  // (sum_i xv_i (x) a1_i)(16x1024) @ Wu_vv'(1024x64):  k = u*32 + v
  for (int kc = 0; kc < 32; ++kc) {
    const float* xu = x + 64 + kc * 3;
    v16bf afr;
#pragma unroll
    for (int j = 0; j < 16; ++j) {
      int v = aK(half, j);
      const float* av = a + 64 + v * 3;
      afr[j] = (__bf16)(xu[0] * av[0] + xu[1] * av[1] + xu[2] * av[2]);
    }
#pragma unroll
    for (int nc = 0; nc < 4; ++nc)
      u0c[nc] = wmma_bf16(afr, ldB(Wuvv, 4, kc, nc, L), u0c[nc]);
  }
  // + xs @ Lu_s'
#pragma unroll
  for (int kc = 0; kc < 2; ++kc) {
    v16bf afr;
#pragma unroll
    for (int j = 0; j < 16; ++j) afr[j] = (__bf16)x[kc * 32 + aK(half, j)];
#pragma unroll
    for (int nc = 0; nc < 4; ++nc)
      u0c[nc] = wmma_bf16(afr, ldB(Lus, 4, kc, nc, L), u0c[nc]);
  }
  // write u0 (residual add) early to lower register pressure
  {
    const int ncol = L & 15;
#pragma unroll
    for (int nc = 0; nc < 4; ++nc) {
      int ch = nc * 16 + ncol;
#pragma unroll
      for (int j = 0; j < 8; ++j) {
        int m = half * 8 + j;
        out[(n0 + m) * NF_DIM + ch] = lds_x[w][m][ch] + u0c[nc][j];
      }
    }
  }

  // ---------- u1 (per vector component i) ----------
  for (int i = 0; i < 3; ++i) {
    v8f u1c[2];
    u1c[0] = vzero();
    u1c[1] = vzero();
    // (xs (x) a1_i)(16x2048) @ Wu_sv'(2048x32):  k = u*32 + v  (u = kc)
    for (int kc = 0; kc < 64; ++kc) {
      const float xu = x[kc];
      v16bf afr;
#pragma unroll
      for (int j = 0; j < 16; ++j) {
        int v = aK(half, j);
        afr[j] = (__bf16)(xu * a[64 + v * 3 + i]);
      }
#pragma unroll
      for (int nc = 0; nc < 2; ++nc)
        u1c[nc] = wmma_bf16(afr, ldB(Wusv, 2, kc, nc, L), u1c[nc]);
    }
    // (xv_i (x) a0)(16x2048) @ Wu_vs'(2048x32):  k = u*64 + v
    for (int kc = 0; kc < 64; ++kc) {
      const int u = kc >> 1;
      const int vb = (kc & 1) << 5;
      const float xu = x[64 + u * 3 + i];
      v16bf afr;
#pragma unroll
      for (int j = 0; j < 16; ++j) {
        int v = vb + aK(half, j);
        afr[j] = (__bf16)(xu * a[v]);
      }
#pragma unroll
      for (int nc = 0; nc < 2; ++nc)
        u1c[nc] = wmma_bf16(afr, ldB(Wuvs, 2, kc, nc, L), u1c[nc]);
    }
    // + xv_i @ Lu_v'
    {
      v16bf afr;
#pragma unroll
      for (int j = 0; j < 16; ++j) {
        int k = aK(half, j);
        afr[j] = (__bf16)x[64 + k * 3 + i];
      }
#pragma unroll
      for (int nc = 0; nc < 2; ++nc)
        u1c[nc] = wmma_bf16(afr, ldB(Luv, 2, 0, nc, L), u1c[nc]);
    }
    // write u1_i (residual add): channel 64 + w*3 + i
    const int ncol = L & 15;
#pragma unroll
    for (int nc = 0; nc < 2; ++nc) {
      int ch = 64 + (nc * 16 + ncol) * 3 + i;
#pragma unroll
      for (int j = 0; j < 8; ++j) {
        int m = half * 8 + j;
        out[(n0 + m) * NF_DIM + ch] = lds_x[w][m][ch] + u1c[nc][j];
      }
    }
  }
}

// ---------------- host launch ----------------
extern "C" void kernel_launch(void* const* d_in, const int* in_sizes, int n_in,
                              void* d_out, int out_size, void* d_ws,
                              size_t ws_size, hipStream_t stream) {
  const float* nf    = (const float*)d_in[0];
  const int*   ei    = (const int*)d_in[1];
  const float* ea    = (const float*)d_in[2];
  // d_in[3] node_attr_scalar_raw: unused by the reference
  const float* Wm_ss = (const float*)d_in[4];
  const float* Wm_vv = (const float*)d_in[5];
  const float* Wm_sv = (const float*)d_in[6];
  const float* Wm_vs = (const float*)d_in[7];
  const float* Lm_s  = (const float*)d_in[8];
  const float* Lm_v  = (const float*)d_in[9];
  const float* Wu_ss = (const float*)d_in[10];
  const float* Wu_vv = (const float*)d_in[11];
  const float* Wu_sv = (const float*)d_in[12];
  const float* Wu_vs = (const float*)d_in[13];
  const float* Lu_s  = (const float*)d_in[14];
  const float* Lu_v  = (const float*)d_in[15];

  const int Nn = in_sizes[0] / NF_DIM;  // 20000
  const int En = in_sizes[2] / 4;       // 640000

  float* agg = (float*)d_ws;
  size_t aggBytes = ((size_t)Nn * NF_DIM * sizeof(float) + 255) & ~(size_t)255;
  __bf16* wb = (__bf16*)((char*)d_ws + aggBytes);

  // zero the aggregation buffer
  {
    int n4 = Nn * NF_DIM / 4;
    int grid = (n4 + 255) / 256;
    if (grid > 4096) grid = 4096;
    zero_kernel<<<grid, 256, 0, stream>>>((float4*)agg, n4);
  }

  // weight prep: bf16 cast + constant folding + B-fragment swizzle
  const float IS3   = 0.5773502691896258f;   // 1/sqrt(3)
  const float I8    = 0.125f;
  const float IS32  = 0.17677669529663687f;  // 1/sqrt(32)
  const float I64   = 0.015625f;
  const float I32   = 0.03125f;
  const float IS2048= 0.022097086912079608f; // 1/sqrt(2048)

  auto prep = [&](const float* src, int off, int KC, int NC, int Ntot,
                  float scale) {
    int total = KC * NC * 512;
    int grid = (total + 255) / 256;
    prep_weight_kernel<<<grid, 256, 0, stream>>>(src, wb + off, KC, NC, Ntot,
                                                 scale);
  };
  prep(Wm_ss, OFF_WMSS, 2, 4, 64, I8);          // A carries s*dist
  prep(Lm_s,  OFF_LMS,  2, 4, 64, I8);
  prep(Wm_vv, OFF_WMVV, 1, 4, 64, IS3 * IS32);
  prep(Wm_sv, OFF_WMSV, 2, 2, 32, I8);          // r_vec applied post-WMMA
  prep(Wm_vs, OFF_WMVS, 1, 2, 32, IS32);        // A carries v*dist
  prep(Lm_v,  OFF_LMV,  1, 2, 32, IS32);
  prep(Wu_ss, OFF_WUSS, 128, 4, 64, I64);
  prep(Wu_vv, OFF_WUVV, 32, 4, 64, IS3 * I32);
  prep(Wu_sv, OFF_WUSV, 64, 2, 32, IS2048);
  prep(Wu_vs, OFF_WUVS, 64, 2, 32, IS2048);
  prep(Lu_s,  OFF_LUS,  2, 4, 64, I8);
  prep(Lu_v,  OFF_LUV,  1, 2, 32, IS32);

  // edge / message phase
  {
    int tiles = (En + 15) / 16;
    int grid = (tiles + EW - 1) / EW;
    edge_kernel<<<grid, EW * 32, 0, stream>>>(nf, ei, ea, agg, wb, En);
  }
  // node / update phase
  {
    int tiles = (Nn + 15) / 16;
    int grid = (tiles + NW - 1) / NW;
    node_kernel<<<grid, NW * 32, 0, stream>>>(nf, agg, wb, (float*)d_out, Nn);
  }
}
